// ARCDModelPTADisc_712964571500
// MI455X (gfx1250) — compile-verified, matrix operand order checked by reading, not run on hardware
//
#include <hip/hip_runtime.h>

#define D_    256
#define DIN_  128
#define NDOM  512
#define NSK   2048
#define NQ    16384
#define NU    8192
#define EPSF  1e-5f

// ---------------------------------------------------------------------------
// WMMA fragment types (gfx1250 wave32: 16x16x32 bf16 -> f32)
// ---------------------------------------------------------------------------
typedef __attribute__((ext_vector_type(16))) __bf16 bf16x16;
typedef __attribute__((ext_vector_type(8)))  float  floatx8;

union FragB { bf16x16 v; unsigned short u[16]; uint4 q[2]; };
union FragC { floatx8  v; float          f[8]; };

__device__ __forceinline__ unsigned short f2bf(float f) {
  unsigned int u = __float_as_uint(f);
  unsigned int r = u + 0x7FFFu + ((u >> 16) & 1u);   // round-to-nearest-even
  return (unsigned short)(r >> 16);
}
__device__ __forceinline__ unsigned int pk2bf(float a, float b) {
  return (unsigned int)f2bf(a) | ((unsigned int)f2bf(b) << 16);
}

// ---------------------------------------------------------------------------
// Generic WMMA GEMM:
//   C[m,n] = epi( sum_k  scaleA(A(m,k)) * bf16(Bt[n,k]) )
//   scaleA(a) = (a + (DIAG && m==k)) * rs[m] * cs[k]      (template-selected)
//   epi(x)    = maybe_relu(x + bias[n]) + addm[m,n]
// A fp32 (optionally transposed access), Bt bf16 row-major [N,K].
//
// Block = 256 threads = 8 waves. Block tile 64(M) x 256(N): one block spans
// the whole N so each fp32 A element is fetched from global and converted to
// bf16 exactly once, staged in (double-buffered) LDS, and re-read by all 8
// waves via ds_load_b128. Wave tile 64x32 (8 accumulators) halves B re-read
// traffic vs a 32-row tile. B (bf16) comes straight from global / L2.
// Requires M%64==0, N%256==0, K%32==0 (true for all calls here).
// ---------------------------------------------------------------------------
template<bool TRANSA, bool DIAG, bool HASRS, bool HASCS>
__global__ void __launch_bounds__(256)
gemm_wmma_kernel(const float* __restrict__ A, int lda,
                 const unsigned short* __restrict__ Bt,
                 const float* __restrict__ rs,
                 const float* __restrict__ cs,
                 const float* __restrict__ bias,
                 const float* __restrict__ addm,
                 float* __restrict__ C,
                 int M, int N, int K, int doRelu)
{
  __shared__ unsigned short As[2][64 * 32];   // 2 x 4KB, [row][k] bf16

  const int tid  = threadIdx.x;
  const int lane = tid & 31;
  const int wave = tid >> 5;                  // 0..7 -> 32-col slice
  const int mW   = blockIdx.y * 64;
  const int nW   = blockIdx.x * 256 + wave * 32;
  const int l    = lane & 15;
  const int hi   = lane >> 4;

  FragC c[4][2];
#pragma unroll
  for (int i = 0; i < 4; ++i)
#pragma unroll
    for (int j = 0; j < 2; ++j)
#pragma unroll
      for (int r = 0; r < 8; ++r) c[i][j].f[r] = 0.f;

  const unsigned short* pB0 = Bt + (size_t)(nW + l)      * K;
  const unsigned short* pB1 = Bt + (size_t)(nW + 16 + l) * K;

  // staging decomposition of the 64x32 A tile (2048 elems / 256 threads = 8)
  const int srow = tid >> 2;           // 0..63  (!TRANSA: tile row)
  const int skq  = (tid & 3) * 8;      // 0..24  (!TRANSA: k oct)
  const int scol = tid >> 3;           // 0..31  (TRANSA: tile k)
  const int srq  = (tid & 7) * 8;      // 0..56  (TRANSA: row oct)

  int buf = 0;
  for (int k0 = 0; k0 < K; k0 += 32, buf ^= 1) {
    // ---- cooperative stage: fp32 -> scaled bf16 tile in LDS ----
    if (!TRANSA) {
      const int grow = mW + srow;
      const int gk   = k0 + skq;
      const float4 f0 = *(const float4*)(A + (size_t)grow * lda + gk);
      const float4 f1 = *(const float4*)(A + (size_t)grow * lda + gk + 4);
      float t[8] = { f0.x, f0.y, f0.z, f0.w, f1.x, f1.y, f1.z, f1.w };
      if (DIAG) {
        const int dj = grow - gk;
        if (dj >= 0 && dj < 8) t[dj] += 1.0f;
      }
      if (HASCS) {
        const float4 s0 = *(const float4*)(cs + gk);
        const float4 s1 = *(const float4*)(cs + gk + 4);
        t[0] *= s0.x; t[1] *= s0.y; t[2] *= s0.z; t[3] *= s0.w;
        t[4] *= s1.x; t[5] *= s1.y; t[6] *= s1.z; t[7] *= s1.w;
      }
      const float rv = HASRS ? rs[grow] : 1.0f;
      uint4 o;
      o.x = pk2bf(t[0] * rv, t[1] * rv);
      o.y = pk2bf(t[2] * rv, t[3] * rv);
      o.z = pk2bf(t[4] * rv, t[5] * rv);
      o.w = pk2bf(t[6] * rv, t[7] * rv);
      *(uint4*)(&As[buf][srow * 32 + skq]) = o;
    } else {
      // transposed A: read 8 consecutive rows (coalesced), transpose via LDS
      const int gk   = k0 + scol;
      const int grow = mW + srq;
      const float4 f0 = *(const float4*)(A + (size_t)gk * lda + grow);
      const float4 f1 = *(const float4*)(A + (size_t)gk * lda + grow + 4);
      float t[8] = { f0.x, f0.y, f0.z, f0.w, f1.x, f1.y, f1.z, f1.w };
      float r[8] = { 1.f, 1.f, 1.f, 1.f, 1.f, 1.f, 1.f, 1.f };
      if (HASRS) {
        const float4 rv0 = *(const float4*)(rs + grow);
        const float4 rv1 = *(const float4*)(rs + grow + 4);
        r[0] = rv0.x; r[1] = rv0.y; r[2] = rv0.z; r[3] = rv0.w;
        r[4] = rv1.x; r[5] = rv1.y; r[6] = rv1.z; r[7] = rv1.w;
      }
#pragma unroll
      for (int j = 0; j < 8; ++j)
        As[buf][(srq + j) * 32 + scol] = f2bf(t[j] * r[j]);
    }
    __syncthreads();

    // ---- fragments ----
    FragB b0, b1;
    b0.q[0] = *(const uint4*)(pB0 + k0 + hi * 8);
    b0.q[1] = *(const uint4*)(pB0 + k0 + 16 + hi * 8);
    b1.q[0] = *(const uint4*)(pB1 + k0 + hi * 8);
    b1.q[1] = *(const uint4*)(pB1 + k0 + 16 + hi * 8);

#pragma unroll
    for (int i = 0; i < 4; ++i) {
      FragB a;
      a.q[0] = *(const uint4*)(&As[buf][(i * 16 + l) * 32 + hi * 8]);
      a.q[1] = *(const uint4*)(&As[buf][(i * 16 + l) * 32 + 16 + hi * 8]);
      c[i][0].v = __builtin_amdgcn_wmma_f32_16x16x32_bf16(false, a.v, false, b0.v, (short)0, c[i][0].v, false, false);
      c[i][1].v = __builtin_amdgcn_wmma_f32_16x16x32_bf16(false, a.v, false, b1.v, (short)0, c[i][1].v, false, false);
    }
    // next iteration stages into the other buffer; the barrier above keeps
    // any re-stage of this buffer two iterations away from these reads
  }

  auto storeTile = [&](FragC& cf, int mT, int nT) {
    int   col = nT + l;
    float bv  = bias ? bias[col] : 0.0f;
#pragma unroll
    for (int r = 0; r < 8; ++r) {
      int   row = mT + hi * 8 + r;
      float v   = cf.f[r] + bv;
      if (doRelu) v = fmaxf(v, 0.0f);
      if (addm)   v += addm[(size_t)row * N + col];
      C[(size_t)row * N + col] = v;
    }
  };
#pragma unroll
  for (int i = 0; i < 4; ++i) {
    storeTile(c[i][0], mW + i * 16, nW);
    storeTile(c[i][1], mW + i * 16, nW + 16);
  }
}

// ---------------------------------------------------------------------------
// Row sums -> scale factors.  mode 0: rsqrt(sum+1) (GCN D^-1/2 with self-loop)
//                             mode 1: 1/max(sum,1) (bipartite degree)
// ---------------------------------------------------------------------------
__global__ void rowsum_kernel(const float* __restrict__ A, int K,
                              float* __restrict__ out, int mode)
{
  __shared__ float red[256];
  int row = blockIdx.x;
  const float* p = A + (size_t)row * K;
  float s = 0.f;
  for (int k = threadIdx.x; k < K; k += 256) s += p[k];
  red[threadIdx.x] = s; __syncthreads();
  for (int off = 128; off > 0; off >>= 1) {
    if (threadIdx.x < off) red[threadIdx.x] += red[threadIdx.x + off];
    __syncthreads();
  }
  if (threadIdx.x == 0) {
    float v = red[0];
    out[row] = (mode == 0) ? rsqrtf(v + 1.0f) : 1.0f / fmaxf(v, 1.0f);
  }
}

__global__ void zero_kernel(float* __restrict__ p, int n)
{
  int i = blockIdx.x * blockDim.x + threadIdx.x;
  if (i < n) p[i] = 0.f;
}

// Partial column sums over a row stripe, accumulated with float atomics.
__global__ void colsum_partial_kernel(const float* __restrict__ A, int Ccols,
                                      int rowsPer, float* __restrict__ acc)
{
  int c  = blockIdx.x * blockDim.x + threadIdx.x;
  int r0 = blockIdx.y * rowsPer;
  float s = 0.f;
  for (int r = r0; r < r0 + rowsPer; ++r) s += A[(size_t)r * Ccols + c];
  atomicAdd(&acc[c], s);
}

__global__ void recip_clamp_kernel(float* __restrict__ p, int n)
{
  int i = blockIdx.x * blockDim.x + threadIdx.x;
  if (i < n) p[i] = 1.0f / fmaxf(p[i], 1.0f);
}

// ---------------------------------------------------------------------------
// fp32 -> bf16 flat convert (weights [N,K] row-major used directly as Bt)
// ---------------------------------------------------------------------------
__global__ void cvt_bf16_kernel(const float* __restrict__ in,
                                unsigned short* __restrict__ out, int n)
{
  int i = blockIdx.x * blockDim.x + threadIdx.x;
  if (i < n) out[i] = f2bf(in[i]);
}

// fp32 [R,C] -> bf16 [C,R] via LDS 32x32 tiles (R,C multiples of 32)
__global__ void transpose_cvt_kernel(const float* __restrict__ in,
                                     unsigned short* __restrict__ out,
                                     int R, int C)
{
  __shared__ float tile[32][33];
  int bx = blockIdx.x * 32;   // column base of input
  int by = blockIdx.y * 32;   // row base of input
  int tx = threadIdx.x;       // 0..31
  int ty = threadIdx.y;       // 0..7
  for (int i = ty; i < 32; i += 8)
    tile[i][tx] = in[(size_t)(by + i) * C + (bx + tx)];
  __syncthreads();
  for (int i = ty; i < 32; i += 8)
    out[(size_t)(bx + i) * R + (by + tx)] = f2bf(tile[tx][i]);
}

// ---------------------------------------------------------------------------
// GAT helpers: sl[n,h] = dot(Wh[n, h*64: ], a_l[h]) ; same for sr
// ---------------------------------------------------------------------------
__global__ void slsr_kernel(const float* __restrict__ Wh,
                            const float* __restrict__ al,
                            const float* __restrict__ ar,
                            float* __restrict__ sl, float* __restrict__ sr)
{
  __shared__ float rl[256], rr[256];
  int n = blockIdx.x, t = threadIdx.x;
  float w = Wh[(size_t)n * D_ + t];
  rl[t] = w * al[t]; rr[t] = w * ar[t];
  __syncthreads();
  for (int off = 32; off > 0; off >>= 1) {
    if ((t & 63) < off) { rl[t] += rl[t + off]; rr[t] += rr[t + off]; }
    __syncthreads();
  }
  if ((t & 63) == 0) {
    int h = t >> 6;
    sl[n * 4 + h] = rl[t];
    sr[n * 4 + h] = rr[t];
  }
}

// Streaming masked-softmax attention + ELU. Block = (n, h), 64 lanes = dk.
__global__ void attn_kernel(const float* __restrict__ Wh,
                            const float* __restrict__ Aadj,
                            const float* __restrict__ sl,
                            const float* __restrict__ sr,
                            float* __restrict__ out, int Nn)
{
  int n = blockIdx.x, h = blockIdx.y, k = threadIdx.x;
  float slv = sl[n * 4 + h];
  float run_m = -3.0e38f, run_s = 0.f, acc = 0.f;
  for (int m = 0; m < Nn; ++m) {
    float am = Aadj[(size_t)n * Nn + m];
    if (am > 0.f || m == n) {                       // mask = (A + I) > 0
      float e = slv + sr[m * 4 + h];
      e = (e > 0.f) ? e : 0.2f * e;                 // leaky_relu 0.2
      float nm = fmaxf(run_m, e);
      float sc = __expf(run_m - nm);
      float p  = __expf(e - nm);
      run_s = run_s * sc + p;
      acc   = acc * sc + p * Wh[(size_t)m * D_ + h * 64 + k];
      run_m = nm;
    }
  }
  float o = (run_s > 0.f) ? acc / run_s : 0.f;      // nan_to_num(0)
  o = (o > 0.f) ? o : (__expf(o) - 1.0f);           // ELU
  out[(size_t)n * D_ + h * 64 + k] = o;
}

// ---------------------------------------------------------------------------
// LayerNorm over D_=256 of (x [+ y]); one block per row
// ---------------------------------------------------------------------------
__global__ void ln_kernel(const float* __restrict__ x,
                          const float* __restrict__ y,
                          const float* __restrict__ g,
                          const float* __restrict__ b,
                          float* __restrict__ out)
{
  __shared__ float red[256];
  int n = blockIdx.x, t = threadIdx.x;
  float v = x[(size_t)n * D_ + t];
  if (y) v += y[(size_t)n * D_ + t];
  red[t] = v; __syncthreads();
  for (int off = 128; off > 0; off >>= 1) {
    if (t < off) red[t] += red[t + off];
    __syncthreads();
  }
  float mean = red[0] * (1.0f / D_); __syncthreads();
  float d = v - mean;
  red[t] = d * d; __syncthreads();
  for (int off = 128; off > 0; off >>= 1) {
    if (t < off) red[t] += red[t + off];
    __syncthreads();
  }
  float var = red[0] * (1.0f / D_);
  out[(size_t)n * D_ + t] = d * rsqrtf(var + EPSF) * g[t] + b[t];
}

// ---------------------------------------------------------------------------
// BatchNorm batch statistics (biased var): parallel partial sums + finalize
// ---------------------------------------------------------------------------
__global__ void bn_partial_kernel(const float* __restrict__ x, int rowsPer,
                                  float* __restrict__ sum, float* __restrict__ sum2)
{
  int c  = threadIdx.x;           // 256 columns
  int r0 = blockIdx.x * rowsPer;
  float s = 0.f, s2 = 0.f;
  for (int r = r0; r < r0 + rowsPer; ++r) {
    float v = x[(size_t)r * D_ + c];
    s += v; s2 += v * v;
  }
  atomicAdd(&sum[c], s);
  atomicAdd(&sum2[c], s2);
}

__global__ void bn_finalize_kernel(const float* __restrict__ sum,
                                   const float* __restrict__ sum2, int Mrows,
                                   float* __restrict__ mean, float* __restrict__ rstd)
{
  int c = threadIdx.x;
  float m   = sum[c] / (float)Mrows;
  float var = sum2[c] / (float)Mrows - m * m;
  mean[c] = m;
  rstd[c] = rsqrtf(var + EPSF);
}

__global__ void bn_apply_kernel(const float* __restrict__ x,
                                const float* __restrict__ mean,
                                const float* __restrict__ rstd,
                                const float* __restrict__ g,
                                const float* __restrict__ b,
                                float* __restrict__ out, int n)
{
  int i = blockIdx.x * blockDim.x + threadIdx.x;
  if (i >= n) return;
  int c = i & (D_ - 1);
  out[i] = (x[i] - mean[c]) * rstd[c] * g[c] + b[c];
}

// ---------------------------------------------------------------------------
// Host-side helpers
// ---------------------------------------------------------------------------
static inline void launch_gemm(hipStream_t s, const float* A, int lda,
                               const unsigned short* Bt, const float* rs,
                               const float* cs, int addDiag, const float* bias,
                               const float* addm, float* C, int M, int N, int K,
                               int relu, bool transA)
{
  dim3 grid(N / 256, M / 64), block(256);
  if (transA)        // transposed A with row scaling (bipartite backward)
    gemm_wmma_kernel<true,  false, true,  false><<<grid, block, 0, s>>>(A, lda, Bt, rs, cs, bias, addm, C, M, N, K, relu);
  else if (addDiag)  // symmetric GCN normalization: (A+I) with D^-1/2 both sides
    gemm_wmma_kernel<false, true,  true,  true ><<<grid, block, 0, s>>>(A, lda, Bt, rs, cs, bias, addm, C, M, N, K, relu);
  else if (rs)       // row-scaled bipartite aggregation
    gemm_wmma_kernel<false, false, true,  false><<<grid, block, 0, s>>>(A, lda, Bt, rs, cs, bias, addm, C, M, N, K, relu);
  else               // plain feature x weight
    gemm_wmma_kernel<false, false, false, false><<<grid, block, 0, s>>>(A, lda, Bt, rs, cs, bias, addm, C, M, N, K, relu);
}

static inline void cvt(hipStream_t s, const float* in, unsigned short* out, int n) {
  cvt_bf16_kernel<<<(n + 255) / 256, 256, 0, s>>>(in, out, n);
}
static inline void tr_cvt(hipStream_t s, const float* in, unsigned short* out, int R, int C) {
  transpose_cvt_kernel<<<dim3(C / 32, R / 32), dim3(32, 8), 0, s>>>(in, out, R, C);
}

extern "C" void kernel_launch(void* const* d_in, const int* in_sizes, int n_in,
                              void* d_out, int out_size, void* d_ws, size_t ws_size,
                              hipStream_t stream)
{
  (void)in_sizes; (void)n_in; (void)out_size; (void)ws_size;
  // -------- inputs (setup_inputs order, params dict flattened) --------
  const float* H_s    = (const float*)d_in[0];   // [2048,128]
  const float* H_d    = (const float*)d_in[1];   // [512,128]
  const float* A_dom  = (const float*)d_in[2];   // [512,512]
  const float* A_ds   = (const float*)d_in[3];   // [2048,512]
  const float* A_pre  = (const float*)d_in[4];   // [2048,2048]
  const float* A_qs   = (const float*)d_in[5];   // [16384,2048]
  const float* A_uq   = (const float*)d_in[6];   // [8192,16384]
  const float* db_W   = (const float*)d_in[7];
  const float* db_b   = (const float*)d_in[8];
  const float* da_W   = (const float*)d_in[9];
  const float* da_al  = (const float*)d_in[10];
  const float* da_ar  = (const float*)d_in[11];
  const float* d_ln_g = (const float*)d_in[12];
  const float* d_ln_b = (const float*)d_in[13];
  const float* d_proj_W = (const float*)d_in[14];
  const float* d_proj_b = (const float*)d_in[15];
  const float* d2s_W  = (const float*)d_in[16];
  const float* d2s_b  = (const float*)d_in[17];
  const float* sb_W   = (const float*)d_in[18];
  const float* sb_b   = (const float*)d_in[19];
  const float* sa_W   = (const float*)d_in[20];
  const float* sa_al  = (const float*)d_in[21];
  const float* sa_ar  = (const float*)d_in[22];
  const float* s_ln_g = (const float*)d_in[23];
  const float* s_ln_b = (const float*)d_in[24];
  const float* s_proj_W = (const float*)d_in[25];
  const float* s_proj_b = (const float*)d_in[26];
  const float* m_ln_g = (const float*)d_in[27];
  const float* m_ln_b = (const float*)d_in[28];
  const float* q_emb  = (const float*)d_in[29];  // [16384,256]
  const float* u_emb  = (const float*)d_in[30];  // [8192,256]
  const float* su_W   = (const float*)d_in[31];
  const float* su_b   = (const float*)d_in[32];
  const float* u_ln_g = (const float*)d_in[33];
  const float* u_ln_b = (const float*)d_in[34];
  const float* qf0_W  = (const float*)d_in[35];
  const float* qf0_b  = (const float*)d_in[36];
  const float* qb0_W  = (const float*)d_in[37];
  const float* qb0_b  = (const float*)d_in[38];
  const float* bn0_g  = (const float*)d_in[39];
  const float* bn0_b  = (const float*)d_in[40];
  const float* qf1_W  = (const float*)d_in[41];
  const float* qf1_b  = (const float*)d_in[42];
  // d_in[43], d_in[44] = qb1_W / qb1_b : dead in the reference (H_src unused)
  const float* bn1_g  = (const float*)d_in[45];
  const float* bn1_b  = (const float*)d_in[46];

  // -------- outputs (concatenated) --------
  float* out_hs = (float*)d_out;                 // [2048,256]
  float* out_hq = out_hs + (size_t)NSK * D_;     // [16384,256]
  float* out_hu = out_hq + (size_t)NQ  * D_;     // [8192,256]

  // -------- workspace layout --------
  char*  wsb = (char*)d_ws;
  size_t off = 0;
  auto wsalloc = [&](size_t bytes) -> void* {
    void* p = wsb + off;
    off = (off + bytes + 255) & ~(size_t)255;
    return p;
  };
  float* F1  = (float*)wsalloc((size_t)NQ * D_ * 4);
  float* F2  = (float*)wsalloc((size_t)NQ * D_ * 4);
  float* F3  = (float*)wsalloc((size_t)NQ * D_ * 4);
  float* HT  = (float*)wsalloc((size_t)NQ * D_ * 4);
  unsigned short* BT = (unsigned short*)wsalloc((size_t)D_ * NQ * 2);
  unsigned short* WB = (unsigned short*)wsalloc((size_t)D_ * D_ * 2);
  float* HD   = (float*)wsalloc((size_t)NDOM * D_ * 4);
  float* HD2S = (float*)wsalloc((size_t)NSK  * D_ * 4);
  float* SL   = (float*)wsalloc((size_t)NSK * 4 * 4);
  float* SR   = (float*)wsalloc((size_t)NSK * 4 * 4);
  float* RSD  = (float*)wsalloc(NDOM * 4);
  float* RDS  = (float*)wsalloc(NSK * 4);
  float* RSS  = (float*)wsalloc(NSK * 4);
  float* RQ   = (float*)wsalloc(NQ * 4);
  float* RQT  = (float*)wsalloc(NSK * 4);
  float* RU   = (float*)wsalloc(NU * 4);
  float* MEAN = (float*)wsalloc(D_ * 4);
  float* RSTD = (float*)wsalloc(D_ * 4);
  float* SUM  = (float*)wsalloc(D_ * 4);
  float* SUM2 = (float*)wsalloc(D_ * 4);

  auto bn_stats = [&](const float* x, int Mrows) {
    zero_kernel<<<1, 256, 0, stream>>>(SUM, D_);
    zero_kernel<<<1, 256, 0, stream>>>(SUM2, D_);
    bn_partial_kernel<<<Mrows / 256, 256, 0, stream>>>(x, 256, SUM, SUM2);
    bn_finalize_kernel<<<1, 256, 0, stream>>>(SUM, SUM2, Mrows, MEAN, RSTD);
  };

  // ===================== DomainGCN =====================
  rowsum_kernel<<<NDOM, 256, 0, stream>>>(A_dom, NDOM, RSD, 0);        // D^-1/2
  cvt(stream, db_W, WB, D_ * DIN_);
  launch_gemm(stream, H_d, DIN_, WB, 0, 0, 0, db_b, 0, F1, NDOM, D_, DIN_, 0, false);   // HW+b
  tr_cvt(stream, F1, BT, NDOM, D_);                                    // (HW)^T bf16
  launch_gemm(stream, A_dom, NDOM, BT, RSD, RSD, 1, 0, 0, F2, NDOM, D_, NDOM, 1, false); // hd0
  cvt(stream, da_W, WB, D_ * D_);
  launch_gemm(stream, F2, D_, WB, 0, 0, 0, 0, 0, F3, NDOM, D_, D_, 0, false);           // Wh_d
  slsr_kernel<<<NDOM, 256, 0, stream>>>(F3, da_al, da_ar, SL, SR);
  attn_kernel<<<dim3(NDOM, 4), 64, 0, stream>>>(F3, A_dom, SL, SR, F1, NDOM);           // attn (ELU)
  cvt(stream, d_proj_W, WB, D_ * DIN_);
  launch_gemm(stream, H_d, DIN_, WB, 0, 0, 0, d_proj_b, F1, F2, NDOM, D_, DIN_, 0, false); // +proj
  ln_kernel<<<NDOM, 256, 0, stream>>>(F2, 0, d_ln_g, d_ln_b, HD);      // hd

  // ---- domain -> skill bipartite ----
  rowsum_kernel<<<NSK, 256, 0, stream>>>(A_ds, NDOM, RDS, 1);
  cvt(stream, d2s_W, WB, D_ * D_);
  launch_gemm(stream, HD, D_, WB, 0, 0, 0, d2s_b, 0, F1, NDOM, D_, D_, 0, false);
  tr_cvt(stream, F1, BT, NDOM, D_);
  launch_gemm(stream, A_ds, NDOM, BT, RDS, 0, 0, 0, 0, HD2S, NSK, D_, NDOM, 1, false);  // h_d2s

  // ===================== SkillGCN =====================
  rowsum_kernel<<<NSK, 256, 0, stream>>>(A_pre, NSK, RSS, 0);
  cvt(stream, sb_W, WB, D_ * DIN_);
  launch_gemm(stream, H_s, DIN_, WB, 0, 0, 0, sb_b, 0, F1, NSK, D_, DIN_, 0, false);
  tr_cvt(stream, F1, BT, NSK, D_);
  launch_gemm(stream, A_pre, NSK, BT, RSS, RSS, 1, 0, 0, F2, NSK, D_, NSK, 1, false);   // hs0
  cvt(stream, sa_W, WB, D_ * D_);
  launch_gemm(stream, F2, D_, WB, 0, 0, 0, 0, 0, F3, NSK, D_, D_, 0, false);            // Wh_s
  slsr_kernel<<<NSK, 256, 0, stream>>>(F3, sa_al, sa_ar, SL, SR);
  attn_kernel<<<dim3(NSK, 4), 64, 0, stream>>>(F3, A_pre, SL, SR, F1, NSK);
  cvt(stream, s_proj_W, WB, D_ * DIN_);
  launch_gemm(stream, H_s, DIN_, WB, 0, 0, 0, s_proj_b, F1, F2, NSK, D_, DIN_, 0, false);
  ln_kernel<<<NSK, 256, 0, stream>>>(F2, 0, s_ln_g, s_ln_b, F3);       // hsk
  ln_kernel<<<NSK, 256, 0, stream>>>(F3, HD2S, m_ln_g, m_ln_b, out_hs); // h_s_out

  // ===================== Question stack =====================
  rowsum_kernel<<<NQ, 256, 0, stream>>>(A_qs, NSK, RQ, 1);             // fwd degrees
  zero_kernel<<<NSK / 256, 256, 0, stream>>>(RQT, NSK);                // bwd degrees
  colsum_partial_kernel<<<dim3(NSK / 256, 32), 256, 0, stream>>>(A_qs, NSK, NQ / 32, RQT);
  recip_clamp_kernel<<<NSK / 256, 256, 0, stream>>>(RQT, NSK);
  // layer 0 forward
  cvt(stream, qf0_W, WB, D_ * D_);
  launch_gemm(stream, out_hs, D_, WB, 0, 0, 0, qf0_b, 0, F1, NSK, D_, D_, 0, false);
  tr_cvt(stream, F1, BT, NSK, D_);
  launch_gemm(stream, A_qs, NSK, BT, RQ, 0, 0, 0, q_emb, F2, NQ, D_, NSK, 1, false);
  bn_stats(F2, NQ);
  bn_apply_kernel<<<(NQ * D_) / 256, 256, 0, stream>>>(F2, MEAN, RSTD, bn0_g, bn0_b, HT, NQ * D_);
  // layer 0 backward (A_qs^T)
  cvt(stream, qb0_W, WB, D_ * D_);
  launch_gemm(stream, HT, D_, WB, 0, 0, 0, qb0_b, 0, F2, NQ, D_, D_, 0, false);
  tr_cvt(stream, F2, BT, NQ, D_);
  launch_gemm(stream, A_qs, NSK, BT, RQT, 0, 0, 0, 0, F1, NSK, D_, NQ, 1, true);        // H_src
  // layer 1 forward
  cvt(stream, qf1_W, WB, D_ * D_);
  launch_gemm(stream, F1, D_, WB, 0, 0, 0, qf1_b, 0, F2, NSK, D_, D_, 0, false);
  tr_cvt(stream, F2, BT, NSK, D_);
  launch_gemm(stream, A_qs, NSK, BT, RQ, 0, 0, 0, HT, F3, NQ, D_, NSK, 1, false);
  bn_stats(F3, NQ);
  bn_apply_kernel<<<(NQ * D_) / 256, 256, 0, stream>>>(F3, MEAN, RSTD, bn1_g, bn1_b, out_hq, NQ * D_);

  // ===================== SparseStudentGCN =====================
  rowsum_kernel<<<NU, 256, 0, stream>>>(A_uq, NQ, RU, 1);
  cvt(stream, su_W, WB, D_ * D_);
  launch_gemm(stream, out_hq, D_, WB, 0, 0, 0, su_b, 0, F1, NQ, D_, D_, 0, false);
  tr_cvt(stream, F1, BT, NQ, D_);
  launch_gemm(stream, A_uq, NQ, BT, RU, 0, 0, 0, u_emb, F2, NU, D_, NQ, 1, false);
  ln_kernel<<<NU, 256, 0, stream>>>(F2, 0, u_ln_g, u_ln_b, out_hu);    // h_u
}
